// TransformerBlock_12575664242959
// MI455X (gfx1250) — matrix operand order repaired and checked by reading, not compiled, not run on hardware
//
#include <hip/hip_runtime.h>
#include <hip/hip_bf16.h>

typedef _Float16 half_t;
typedef __attribute__((ext_vector_type(16))) _Float16 v16h;
typedef __attribute__((ext_vector_type(8)))  float    v8f;
typedef __attribute__((ext_vector_type(4)))  unsigned int u32x4;
typedef int v4i __attribute__((vector_size(16)));

union Frag16 { u32x4 u[2]; v16h h; };

__device__ inline v16h load_frag2(const half_t* p0, const half_t* p1) {
  Frag16 f;
  f.u[0] = *reinterpret_cast<const u32x4*>(p0);
  f.u[1] = *reinterpret_cast<const u32x4*>(p1);
  return f.h;
}

// --- CDNA5 async global->LDS staging (guarded; falls back to plain copy) ---
#if __has_builtin(__builtin_amdgcn_global_load_async_to_lds_b128)
#define HAVE_ASYNC_LDS 1
#endif

__device__ inline void stage_b128(const half_t* g, half_t* l) {
#ifdef HAVE_ASYNC_LDS
  __builtin_amdgcn_global_load_async_to_lds_b128(
      (v4i*)(void*)g, (v4i*)(void*)l, 0, 0);
#else
  *reinterpret_cast<u32x4*>(l) = *reinterpret_cast<const u32x4*>(g);
#endif
}

// wait until at most 2 async ops outstanding (previous tile landed,
// next tile still streaming behind the WMMAs)
__device__ inline void wait_stage_drain2() {
#ifdef HAVE_ASYNC_LDS
#if __has_builtin(__builtin_amdgcn_s_wait_asynccnt)
  __builtin_amdgcn_s_wait_asynccnt(2);
#else
  asm volatile("s_wait_asynccnt 0x2" ::: "memory");
#endif
#endif
}

__device__ inline void wait_stage_all() {
#ifdef HAVE_ASYNC_LDS
#if __has_builtin(__builtin_amdgcn_s_wait_asynccnt)
  __builtin_amdgcn_s_wait_asynccnt(0);
#else
  asm volatile("s_wait_asynccnt 0x0" ::: "memory");
#endif
#endif
}

// ---------------------------------------------------------------------------
// weight conversion / packing (transposed: Bt[N][K] so GEMM staging is linear)
// ---------------------------------------------------------------------------
__global__ void convert_transpose_f16(const float* __restrict__ in,
                                      half_t* __restrict__ out, int R, int C) {
  int i = blockIdx.x * 256 + threadIdx.x;
  if (i >= R * C) return;
  int r = i / C, c = i % C;
  out[(size_t)c * R + r] = (half_t)in[i];
}

// wq/wk/wv [H,E,D] -> Bt [3*H*D][E] f16 (rows: q | k | v, each h*D+d)
__global__ void pack_qkvT_f16(const float* __restrict__ wq,
                              const float* __restrict__ wk,
                              const float* __restrict__ wv,
                              half_t* __restrict__ out,
                              int E, int H, int D) {
  int i = blockIdx.x * 256 + threadIdx.x;
  int N = 3 * H * D;
  if (i >= N * E) return;
  int col = i / E, e = i % E;
  int which = col / (H * D);
  int hd = col % (H * D);
  int h = hd / D, d = hd % D;
  const float* src = (which == 0) ? wq : (which == 1) ? wk : wv;
  out[i] = (half_t)src[((size_t)h * E + e) * D + d];
}

// qkv16 [B*T, 3HD] (v section) -> vT [B,H,D,T] f16
__global__ void transpose_v_f16(const half_t* __restrict__ qkv,
                                half_t* __restrict__ vT,
                                int Bn, int T, int H, int D) {
  int i = blockIdx.x * 256 + threadIdx.x;
  int total = Bn * H * D * T;
  if (i >= total) return;
  int t = i % T;
  int rest = i / T;
  int d = rest % D;
  int bh = rest / D;
  int h = bh % H, b = bh / H;
  vT[i] = qkv[(size_t)(b * T + t) * (3 * H * D) + 2 * H * D + h * D + d];
}

// ---------------------------------------------------------------------------
// layernorm: f32 [rows,E] -> f16 [rows,E]
// ---------------------------------------------------------------------------
__global__ __launch_bounds__(128) void layernorm_f16(
    const float* __restrict__ X, const float* __restrict__ gw,
    const float* __restrict__ bw, half_t* __restrict__ out, int E) {
  __shared__ float red[128];
  int row = blockIdx.x;
  const float* x = X + (size_t)row * E;
  int tid = threadIdx.x;

  float s = 0.f;
  for (int i = tid; i < E; i += 128) s += x[i];
  red[tid] = s; __syncthreads();
  for (int st = 64; st > 0; st >>= 1) {
    if (tid < st) red[tid] += red[tid + st];
    __syncthreads();
  }
  float mu = red[0] / (float)E;
  __syncthreads();

  float v = 0.f;
  for (int i = tid; i < E; i += 128) { float d = x[i] - mu; v += d * d; }
  red[tid] = v; __syncthreads();
  for (int st = 64; st > 0; st >>= 1) {
    if (tid < st) red[tid] += red[tid + st];
    __syncthreads();
  }
  float rstd = rsqrtf(red[0] / (float)E + 1e-5f);

  half_t* o = out + (size_t)row * E;
  for (int i = tid; i < E; i += 128)
    o[i] = (half_t)((x[i] - mu) * rstd * gw[i] + bw[i]);
}

// ---------------------------------------------------------------------------
// WMMA GEMM: C[M,N] = A[M,K](f16,row-major) * Bt[N][K](f16)  (+bias)(+res)(relu)
// block tile 128x128, 8 waves as 4(M)x2(N); each wave 32x64 (8 accums).
// A fragments straight from global; Bt tiles double-buffered in LDS via
// async global->LDS loads pipelined one K-step ahead.
// ---------------------------------------------------------------------------
__global__ __launch_bounds__(256) void gemm_wmma(
    const half_t* __restrict__ A, int lda,
    const half_t* __restrict__ Bt,  // [N][K]
    int M, int N, int K,
    const float* __restrict__ bias,
    const float* __restrict__ residual,
    float* __restrict__ Cf,
    half_t* __restrict__ Ch,
    int relu) {
  __shared__ __align__(16) half_t Bs[2][128][32];  // 16 KB ping-pong

  int tid = threadIdx.x;
  int wave = tid >> 5, lane = tid & 31;
  int wm = wave >> 1, wn = wave & 1;
  int m0 = blockIdx.y * 128;
  int n0 = blockIdx.x * 128;

  v8f acc[2][4] = {};

  int mlane = lane & 15;
  int kb    = (lane >> 4) * 16;  // A fragment K base
  int koff  = (lane >> 4) * 16;  // B fragment K base

  // each thread stages exactly 2 of the 512 8-half chunks of a 128x32 tile
  int n1 = tid >> 2,         c1 = (tid & 3) * 8;
  int n2 = (tid + 256) >> 2, c2 = (tid & 3) * 8;  // (idx+256)&3 == idx&3

  const int nk = K / 32;
  // prologue: issue tile 0 into buffer 0
  stage_b128(&Bt[(size_t)(n0 + n1) * K + c1], &Bs[0][n1][c1]);
  stage_b128(&Bt[(size_t)(n0 + n2) * K + c2], &Bs[0][n2][c2]);

  for (int ki = 0; ki < nk; ++ki) {
    int kk = ki * 32;
    if (ki + 1 < nk) {
      int kn = kk + 32, bufn = (ki + 1) & 1;
      stage_b128(&Bt[(size_t)(n0 + n1) * K + kn + c1], &Bs[bufn][n1][c1]);
      stage_b128(&Bt[(size_t)(n0 + n2) * K + kn + c2], &Bs[bufn][n2][c2]);
      wait_stage_drain2();  // tile ki landed; tile ki+1 still in flight
    } else {
      wait_stage_all();
    }
    __syncthreads();  // all waves' waits done -> full tile ki visible

    if (kk + 32 < K)  // prefetch next A tile rows for this wave
      __builtin_prefetch(&A[(size_t)(m0 + wm * 32 + mlane) * lda + kk + 32], 0, 1);

    // A fragments (2 M-tiles) straight from global
    v16h a[2];
#pragma unroll
    for (int t = 0; t < 2; ++t) {
      const half_t* ar = &A[(size_t)(m0 + wm * 32 + t * 16 + mlane) * lda + kk];
      a[t] = load_frag2(ar + kb, ar + kb + 8);
    }
    // B fragments (4 N-tiles) from LDS
    int buf = ki & 1;
    v16h bfr[4];
#pragma unroll
    for (int j = 0; j < 4; ++j) {
      const half_t* br = &Bs[buf][wn * 64 + j * 16 + mlane][0];
      bfr[j] = load_frag2(br + koff, br + koff + 8);
    }
#pragma unroll
    for (int t = 0; t < 2; ++t)
#pragma unroll
      for (int j = 0; j < 4; ++j)
        acc[t][j] = __builtin_amdgcn_wmma_f32_16x16x32_f16(
            false, a[t], false, bfr[j], (short)0, acc[t][j], false, false);

    __syncthreads();  // buffer 'buf' free before it is re-targeted at ki+2
  }

  // epilogue (C layout: lane -> column, 8 rows per lane)
  int nl = lane & 15;
  int mh = (lane >> 4) * 8;
#pragma unroll
  for (int t = 0; t < 2; ++t) {
#pragma unroll
    for (int j = 0; j < 4; ++j) {
      int col = n0 + wn * 64 + j * 16 + nl;
      float bv = bias ? bias[col] : 0.0f;
#pragma unroll
      for (int r = 0; r < 8; ++r) {
        int row = m0 + wm * 32 + t * 16 + mh + r;
        float v = acc[t][j][r] + bv;
        if (residual) v += residual[(size_t)row * N + col];
        if (relu) v = fmaxf(v, 0.0f);
        if (Cf) Cf[(size_t)row * N + col] = v;
        else    Ch[(size_t)row * N + col] = (half_t)v;
      }
    }
  }
}

// ---------------------------------------------------------------------------
// attention (faithful: multiplicative tril mask -> 0, softmax over full row).
// qkv [B*T, 3HD] f16, vT [B,H,D,T] f16 -> o16 [B*T, HD] f16.
// Block = 4 independent waves (4 query tiles of one (b,h)); no block barriers:
// K/V fragments load straight from global (L2-resident), P re-layout via
// per-wave LDS (same-wave DS ops are in-order).
// ---------------------------------------------------------------------------
__global__ __launch_bounds__(128) void attention_wmma(
    const half_t* __restrict__ qkv, const half_t* __restrict__ vT,
    half_t* __restrict__ o16, int Bn, int T, int H, int D) {
  (void)Bn;
  const int LDQ = 3 * H * D;  // 1152
  __shared__ __align__(16) half_t Pld[4][16][32];  // per-wave P staging

  int tid = threadIdx.x, wave = tid >> 5, lane = tid & 31;
  int ngroup = T / 64;
  int qg = blockIdx.x % ngroup;
  int bh = blockIdx.x / ngroup;
  int b = bh / H, h = bh % H;

  size_t base = (size_t)b * T * LDQ;
  const half_t* Qp = qkv + base + h * D;
  const half_t* Kp = qkv + base + H * D + h * D;
  const half_t* Vt = vT + (size_t)(b * H + h) * D * T;  // [D][T]

  int q0 = qg * 64 + wave * 16;

  int mlane = lane & 15;
  int kb    = (lane >> 4) * 16;
  int nl    = lane & 15;
  int koff  = (lane >> 4) * 16;

  // Q fragments (both D chunks)
  v16h qa[2];
#pragma unroll
  for (int dc = 0; dc < 2; ++dc) {
    const half_t* qr = Qp + (size_t)(q0 + mlane) * LDQ + dc * 32;
    qa[dc] = load_frag2(qr + kb, qr + kb + 8);
  }

  v8f oacc[4] = {};
  float mrun[8], lrun[8];
#pragma unroll
  for (int r = 0; r < 8; ++r) { mrun[r] = -1e30f; lrun[r] = 0.f; }
  const float scale = rsqrtf((float)D);

  for (int kt = 0; kt < T; kt += 32) {
    // K fragments straight from global: b[i] = K[kt + c*16 + nl][dc*32+koff+i]
    v16h bk[2][2];
#pragma unroll
    for (int c = 0; c < 2; ++c)
#pragma unroll
      for (int dc = 0; dc < 2; ++dc) {
        const half_t* kr = Kp + (size_t)(kt + c * 16 + nl) * LDQ + dc * 32 + koff;
        bk[c][dc] = load_frag2(kr, kr + 8);
      }
    // V fragments: b[i] = Vt[j*16+nl][kt+koff+i]
    v16h bv[4];
#pragma unroll
    for (int j = 0; j < 4; ++j) {
      const half_t* vr = Vt + (size_t)(j * 16 + nl) * T + kt + koff;
      bv[j] = load_frag2(vr, vr + 8);
    }
    if (kt + 32 < T) {  // prefetch next key tile rows
      __builtin_prefetch(Kp + (size_t)(kt + 32 + nl) * LDQ + koff, 0, 1);
      __builtin_prefetch(Vt + (size_t)nl * T + kt + 32 + koff, 0, 1);
    }

    // S = Q K^T (two 16-col tiles)
    v8f s[2] = {};
#pragma unroll
    for (int c = 0; c < 2; ++c)
#pragma unroll
      for (int dc = 0; dc < 2; ++dc)
        s[c] = __builtin_amdgcn_wmma_f32_16x16x32_f16(
            false, qa[dc], false, bk[c][dc], (short)0, s[c], false, false);

    // multiplicative mask -> 0, scale, online softmax over full row
    int rowbase = q0 + ((lane >> 4) << 3);
    float mnew[8], alpha[8];
#pragma unroll
    for (int r = 0; r < 8; ++r) {
      int row = rowbase + r;
      float s0 = ((kt + nl)      <= row) ? s[0][r] * scale : 0.0f;
      float s1 = ((kt + 16 + nl) <= row) ? s[1][r] * scale : 0.0f;
      s[0][r] = s0; s[1][r] = s1;
      float t = fmaxf(s0, s1);
      t = fmaxf(t, __shfl_xor(t, 1));
      t = fmaxf(t, __shfl_xor(t, 2));
      t = fmaxf(t, __shfl_xor(t, 4));
      t = fmaxf(t, __shfl_xor(t, 8));
      float mn = fmaxf(mrun[r], t);
      mnew[r] = mn;
      alpha[r] = __expf(mrun[r] - mn);
      mrun[r] = mn;
    }
#pragma unroll
    for (int r = 0; r < 8; ++r) {
      float p0 = __expf(s[0][r] - mnew[r]);
      float p1 = __expf(s[1][r] - mnew[r]);
      float ts = p0 + p1;
      ts += __shfl_xor(ts, 1);
      ts += __shfl_xor(ts, 2);
      ts += __shfl_xor(ts, 4);
      ts += __shfl_xor(ts, 8);
      lrun[r] = lrun[r] * alpha[r] + ts;
      int m = ((lane >> 4) << 3) + r;
      Pld[wave][m][nl]      = (half_t)p0;
      Pld[wave][m][16 + nl] = (half_t)p1;
#pragma unroll
      for (int j = 0; j < 4; ++j) oacc[j][r] *= alpha[r];
    }

    // P(16x32) x V(32x64); same-wave DS store->load is in-order
    v16h pa = load_frag2(&Pld[wave][mlane][kb], &Pld[wave][mlane][kb + 8]);
#pragma unroll
    for (int j = 0; j < 4; ++j)
      oacc[j] = __builtin_amdgcn_wmma_f32_16x16x32_f16(
          false, pa, false, bv[j], (short)0, oacc[j], false, false);
  }

  // normalize and write O
#pragma unroll
  for (int r = 0; r < 8; ++r) {
    float inv = 1.0f / lrun[r];
    int row = b * T + q0 + ((lane >> 4) << 3) + r;
#pragma unroll
    for (int j = 0; j < 4; ++j) {
      o16[(size_t)row * (H * D) + h * D + j * 16 + nl] =
          (half_t)(oacc[j][r] * inv);
    }
  }
}

// ---------------------------------------------------------------------------
// host launch
// ---------------------------------------------------------------------------
extern "C" void kernel_launch(void* const* d_in, const int* in_sizes, int n_in,
                              void* d_out, int out_size, void* d_ws,
                              size_t ws_size, hipStream_t stream) {
  (void)in_sizes; (void)n_in; (void)out_size; (void)ws_size;
  const int B = 4, T = 2048, E = 384, H = 6, D = 64, F = 4 * E;
  const int BT = B * T;
  const int NQKV = 3 * H * D;  // 1152

  const float* x     = (const float*)d_in[0];
  const float* ln1_g = (const float*)d_in[1];
  const float* ln1_b = (const float*)d_in[2];
  const float* ln2_g = (const float*)d_in[3];
  const float* ln2_b = (const float*)d_in[4];
  const float* wq    = (const float*)d_in[5];
  const float* wk    = (const float*)d_in[6];
  const float* wv    = (const float*)d_in[7];
  const float* wo    = (const float*)d_in[8];
  const float* w1    = (const float*)d_in[9];
  const float* b1    = (const float*)d_in[10];
  const float* w2    = (const float*)d_in[11];
  const float* b2    = (const float*)d_in[12];
  float* out = (float*)d_out;

  char* ws = (char*)d_ws;
  size_t off = 0;
  auto alloc = [&](size_t bytes) -> void* {
    void* p = ws + off;
    off = (off + bytes + 255) & ~(size_t)255;
    return p;
  };
  half_t* h16    = (half_t*)alloc((size_t)BT * E * 2);
  half_t* wqkvT  = (half_t*)alloc((size_t)NQKV * E * 2);  // [1152][384]
  half_t* woT    = (half_t*)alloc((size_t)E * E * 2);     // [384][384]
  half_t* w1T    = (half_t*)alloc((size_t)F * E * 2);     // [1536][384]
  half_t* w2T    = (half_t*)alloc((size_t)E * F * 2);     // [384][1536]
  half_t* qkv16  = (half_t*)alloc((size_t)BT * NQKV * 2);
  half_t* vT     = (half_t*)alloc((size_t)BT * E * 2);    // [B,H,D,T]
  half_t* o16    = (half_t*)alloc((size_t)BT * E * 2);
  float*  xattn  = (float*)alloc((size_t)BT * E * 4);
  half_t* h2_16  = (half_t*)alloc((size_t)BT * E * 2);
  half_t* mid16  = (half_t*)alloc((size_t)BT * F * 2);

  // --- weight conversion (pre-transposed), every call ---
  {
    int n = NQKV * E;
    pack_qkvT_f16<<<(n + 255) / 256, 256, 0, stream>>>(wq, wk, wv, wqkvT, E, H, D);
    n = E * E;   // wo flat [384(hd)][384(e)] -> woT [e][hd]
    convert_transpose_f16<<<(n + 255) / 256, 256, 0, stream>>>(wo, woT, E, E);
    n = E * F;   // w1 [384][1536] -> w1T [1536][384]
    convert_transpose_f16<<<(n + 255) / 256, 256, 0, stream>>>(w1, w1T, E, F);
    // w2 [1536][384] -> w2T [384][1536]
    convert_transpose_f16<<<(n + 255) / 256, 256, 0, stream>>>(w2, w2T, F, E);
  }

  // --- ln1 ---
  layernorm_f16<<<BT, 128, 0, stream>>>(x, ln1_g, ln1_b, h16, E);

  // --- qkv: [BT,384] x Bt[1152][384] -> qkv16 f16 ---
  gemm_wmma<<<dim3(NQKV / 128, BT / 128), 256, 0, stream>>>(
      h16, E, wqkvT, BT, NQKV, E, nullptr, nullptr, nullptr, qkv16, 0);

  // --- vT ---
  {
    int n = BT * E;
    transpose_v_f16<<<(n + 255) / 256, 256, 0, stream>>>(qkv16, vT, B, T, H, D);
  }

  // --- attention ---
  attention_wmma<<<(B * H) * (T / 64), 128, 0, stream>>>(qkv16, vT, o16, B, T, H, D);

  // --- out proj + residual x -> xattn f32 ---
  gemm_wmma<<<dim3(E / 128, BT / 128), 256, 0, stream>>>(
      o16, E, woT, BT, E, E, nullptr, x, xattn, nullptr, 0);

  // --- ln2 ---
  layernorm_f16<<<BT, 128, 0, stream>>>(xattn, ln2_g, ln2_b, h2_16, E);

  // --- ffn1 + bias + relu -> mid16 f16 ---
  gemm_wmma<<<dim3(F / 128, BT / 128), 256, 0, stream>>>(
      h2_16, E, w1T, BT, F, E, b1, nullptr, nullptr, mid16, 1);

  // --- ffn2 + bias + residual xattn -> d_out f32 ---
  gemm_wmma<<<dim3(E / 128, BT / 128), 256, 0, stream>>>(
      mid16, F, w2T, BT, E, F, b2, xattn, out, nullptr, 0);
}
